// Attention_63436666962233
// MI455X (gfx1250) — matrix operand order repaired and checked by reading, not compiled, not run on hardware
//
#include <hip/hip_runtime.h>
#include <hip/hip_bf16.h>

#define B_   4
#define S_   4096
#define E_   64
#define IN_  512
#define BS_  (B_ * S_)

typedef __attribute__((ext_vector_type(16))) _Float16 v16h;
typedef __attribute__((ext_vector_type(8)))  _Float16 v8h;
typedef __attribute__((ext_vector_type(8)))  float    v8f;
typedef __attribute__((ext_vector_type(4)))  float    f4;

static __device__ __forceinline__ v16h cat16(v8h a, v8h b) {
  v16h r;
#pragma unroll
  for (int i = 0; i < 8; ++i) { r[i] = a[i]; r[i + 8] = b[i]; }
  return r;
}

// ---------------------------------------------------------------------------
// Transpose + convert the three weight matrices W[512][64] f32 -> Wt[64][512] f16
// so projection B-fragments are two contiguous 16B loads per lane.
// ---------------------------------------------------------------------------
__global__ __launch_bounds__(256) void prep_w_kernel(
    const float* __restrict__ Wq, const float* __restrict__ Wk, const float* __restrict__ Wv,
    _Float16* __restrict__ Wqt, _Float16* __restrict__ Wkt, _Float16* __restrict__ Wvt) {
  int idx = blockIdx.x * 256 + threadIdx.x;       // 3 * 512 * 64 total
  int m = idx / (IN_ * E_);
  int e = idx - m * (IN_ * E_);
  const float* W  = (m == 0) ? Wq  : (m == 1) ? Wk  : Wv;
  _Float16*    Wt = (m == 0) ? Wqt : (m == 1) ? Wkt : Wvt;
  int k = e >> 6;           // row in W  (input dim)
  int n = e & (E_ - 1);     // col in W  (embed dim)
  Wt[(size_t)n * IN_ + k] = (_Float16)W[e];
}

// ---------------------------------------------------------------------------
// Projection: out = x @ W + b via v_wmma_f32_16x16x32_f16.
// Block = 256 threads = 8 waves; each wave owns a 16x64 output tile.
// transpose_out=1 stores V as Vt[batch][embed][seq] for contiguous PV B-frags.
// ---------------------------------------------------------------------------
__global__ __launch_bounds__(256) void proj_kernel(
    const float* __restrict__ x, const _Float16* __restrict__ Wt,
    const float* __restrict__ bias, _Float16* __restrict__ out, int transpose_out) {
  const int wave = threadIdx.x >> 5;
  const int lane = threadIdx.x & 31;
  const int hi = lane >> 4, lo = lane & 15;
  const int mBase = blockIdx.x * 128 + wave * 16;
  const int row = mBase + lo;

  v8f acc[4];
#pragma unroll
  for (int nt = 0; nt < 4; ++nt)
#pragma unroll
    for (int i = 0; i < 8; ++i) acc[nt][i] = 0.0f;

  for (int k0 = 0; k0 < IN_; k0 += 32) {
    // A fragment (16x32 f16): lane L<16 holds row L, K = {0..7, 16..23} of this k-chunk
    const f4* xp = (const f4*)(x + (size_t)row * IN_ + k0 + hi * 8);
    f4 f0 = xp[0], f1 = xp[1], f2 = xp[4], f3 = xp[5];
    v16h a;
#pragma unroll
    for (int i = 0; i < 4; ++i) {
      a[i]      = (_Float16)f0[i];
      a[i + 4]  = (_Float16)f1[i];
      a[i + 8]  = (_Float16)f2[i];
      a[i + 12] = (_Float16)f3[i];
    }
#pragma unroll
    for (int nt = 0; nt < 4; ++nt) {
      // B fragment (32x16 f16): lane holds column nt*16+lo, K = {0..7, 16..23} (+8 for hi)
      const v8h* wp = (const v8h*)(Wt + (size_t)(nt * 16 + lo) * IN_ + k0 + hi * 8);
      v16h b = cat16(wp[0], wp[2]);
      acc[nt] = __builtin_amdgcn_wmma_f32_16x16x32_f16(
          false, a, false, b, (short)0, acc[nt], false, false);
    }
  }

#pragma unroll
  for (int nt = 0; nt < 4; ++nt) {
    const int col = nt * 16 + lo;
    const float bb = bias[col];
#pragma unroll
    for (int r = 0; r < 8; ++r) {
      const int m = mBase + r + 8 * hi;
      const _Float16 val = (_Float16)(acc[nt][r] + bb);
      if (!transpose_out) {
        out[(size_t)m * E_ + col] = val;
      } else {
        const int b = m >> 12;            // batch (S_=4096)
        const int s = m & (S_ - 1);
        out[((size_t)b * E_ + col) * S_ + s] = val;
      }
    }
  }
}

// ---------------------------------------------------------------------------
// Flash attention: each wave processes 16 query rows against all of S.
// KV tile = 64 -> 16 WMMAs per tile vs one butterfly-reduction pass, softmax in
// exp2 domain (v_exp_f32 is base-2). P re-layout via padded LDS (same-wave).
// ---------------------------------------------------------------------------
#define PSTRIDE 72   // halfs; 144B rows -> 16 distinct banks, 16B aligned

__global__ __launch_bounds__(256) void flash_attn_kernel(
    const _Float16* __restrict__ Qh, const _Float16* __restrict__ Kh,
    const _Float16* __restrict__ Vt, float* __restrict__ out) {
  __shared__ __align__(16) _Float16 pshm[8][16 * PSTRIDE];
  const int wave = threadIdx.x >> 5;
  const int lane = threadIdx.x & 31;
  const int hi = lane >> 4, lo = lane & 15;
  const int batch = blockIdx.y;
  const int qBase = blockIdx.x * 128 + wave * 16;

  const _Float16* Qb = Qh + (size_t)batch * S_ * E_;
  const _Float16* Kb = Kh + (size_t)batch * S_ * E_;
  const _Float16* Vb = Vt + (size_t)batch * E_ * S_;   // transposed: [embed][seq]

  // Q A-fragments: 16x64 = two 16x32 chunks, kept in registers for the whole pass
  v16h aQ[2];
  {
    const int row = qBase + lo;
#pragma unroll
    for (int c = 0; c < 2; ++c) {
      const v8h* qp = (const v8h*)(Qb + (size_t)row * E_ + c * 32 + hi * 8);
      aQ[c] = cat16(qp[0], qp[2]);                     // K {0..7} and {16..23}
    }
  }

  v8f o[4];
  float mrow[8], lrow[8];
#pragma unroll
  for (int nt = 0; nt < 4; ++nt)
#pragma unroll
    for (int i = 0; i < 8; ++i) o[nt][i] = 0.0f;
#pragma unroll
  for (int r = 0; r < 8; ++r) { mrow[r] = -1e30f; lrow[r] = 0.0f; }

  v8f zero8;
#pragma unroll
  for (int i = 0; i < 8; ++i) zero8[i] = 0.0f;

  // fold 1/sqrt(64) and log2(e) into one score scale: softmax done with exp2
  const float SCALE2 = 0.125f * 1.44269504088896340736f;

  for (int kv0 = 0; kv0 < S_; kv0 += 64) {
    // ---- scores: S16x64 = Q(16x64) . K_tile(64x64)^T, chained over embed ----
    v8f s[4];
#pragma unroll
    for (int nt = 0; nt < 4; ++nt) {
      const _Float16* kp = Kb + (size_t)(kv0 + nt * 16 + lo) * E_ + hi * 8;
      v16h b0 = cat16(((const v8h*)kp)[0], ((const v8h*)kp)[2]);               // embed 0..31
      v16h b1 = cat16(((const v8h*)(kp + 32))[0], ((const v8h*)(kp + 32))[2]); // embed 32..63
      v8f t = __builtin_amdgcn_wmma_f32_16x16x32_f16(
          false, aQ[0], false, b0, (short)0, zero8, false, false);
      s[nt] = __builtin_amdgcn_wmma_f32_16x16x32_f16(
          false, aQ[1], false, b1, (short)0, t, false, false);
#pragma unroll
      for (int r = 0; r < 8; ++r) s[nt][r] *= SCALE2;
    }

    // ---- online softmax (exp2 domain), one reduction pass per 64 columns ----
#pragma unroll
    for (int r = 0; r < 8; ++r) {
      float t = fmaxf(fmaxf(s[0][r], s[1][r]), fmaxf(s[2][r], s[3][r]));
      t = fmaxf(t, __shfl_xor(t, 1, 32));
      t = fmaxf(t, __shfl_xor(t, 2, 32));
      t = fmaxf(t, __shfl_xor(t, 4, 32));
      t = fmaxf(t, __shfl_xor(t, 8, 32));
      const float mnew  = fmaxf(mrow[r], t);
      const float alpha = exp2f(mrow[r] - mnew);
      float rs = 0.0f;
#pragma unroll
      for (int nt = 0; nt < 4; ++nt) {
        const float p = exp2f(s[nt][r] - mnew);
        s[nt][r] = p;
        rs += p;
      }
      rs += __shfl_xor(rs, 1, 32);
      rs += __shfl_xor(rs, 2, 32);
      rs += __shfl_xor(rs, 4, 32);
      rs += __shfl_xor(rs, 8, 32);
      lrow[r] = lrow[r] * alpha + rs;
      mrow[r] = mnew;
#pragma unroll
      for (int nt = 0; nt < 4; ++nt) o[nt][r] *= alpha;
    }

    // ---- re-layout P (D-fragment) -> two A-fragments through LDS ----
    _Float16* pw = pshm[wave];
#pragma unroll
    for (int nt = 0; nt < 4; ++nt)
#pragma unroll
      for (int r = 0; r < 8; ++r)
        pw[(r + 8 * hi) * PSTRIDE + nt * 16 + lo] = (_Float16)s[nt][r];
    __builtin_amdgcn_wave_barrier();
    asm volatile("s_wait_dscnt 0x0" ::: "memory");
    v16h aP[2];
#pragma unroll
    for (int c = 0; c < 2; ++c) {
      const v8h* pr = (const v8h*)(pw + lo * PSTRIDE + c * 32 + hi * 8);
      aP[c] = cat16(pr[0], pr[2]);
    }
    __builtin_amdgcn_wave_barrier();

    // ---- O += P(16x64) . V_tile(64x64), chained over kv ----
#pragma unroll
    for (int nt = 0; nt < 4; ++nt) {
      const _Float16* vp = Vb + (size_t)(nt * 16 + lo) * S_ + kv0 + hi * 8;
      v16h bv0 = cat16(((const v8h*)vp)[0], ((const v8h*)vp)[2]);               // kv 0..31
      v16h bv1 = cat16(((const v8h*)(vp + 32))[0], ((const v8h*)(vp + 32))[2]); // kv 32..63
      o[nt] = __builtin_amdgcn_wmma_f32_16x16x32_f16(
          false, aP[0], false, bv0, (short)0, o[nt], false, false);
      o[nt] = __builtin_amdgcn_wmma_f32_16x16x32_f16(
          false, aP[1], false, bv1, (short)0, o[nt], false, false);
    }
  }

  // ---- epilogue: normalize (one rcp per row) and store fp32 ----
  float inv[8];
#pragma unroll
  for (int r = 0; r < 8; ++r) inv[r] = 1.0f / lrow[r];
#pragma unroll
  for (int nt = 0; nt < 4; ++nt)
#pragma unroll
    for (int r = 0; r < 8; ++r) {
      const int row = qBase + r + 8 * hi;
      out[((size_t)batch * S_ + row) * E_ + nt * 16 + lo] = o[nt][r] * inv[r];
    }
}

// ---------------------------------------------------------------------------
extern "C" void kernel_launch(void* const* d_in, const int* in_sizes, int n_in,
                              void* d_out, int out_size, void* d_ws, size_t ws_size,
                              hipStream_t stream) {
  const float* q  = (const float*)d_in[0];
  const float* k  = (const float*)d_in[1];
  const float* v  = (const float*)d_in[2];
  const float* Wq = (const float*)d_in[3];
  const float* bq = (const float*)d_in[4];
  const float* Wk = (const float*)d_in[5];
  const float* bk = (const float*)d_in[6];
  const float* Wv = (const float*)d_in[7];
  const float* bv = (const float*)d_in[8];
  float* out = (float*)d_out;

  // workspace layout (f16): Qh | Kh | Vt | Wqt | Wkt | Wvt  (~6.2 MB)
  _Float16* Qh  = (_Float16*)d_ws;
  _Float16* Kh  = Qh  + (size_t)BS_ * E_;
  _Float16* Vt  = Kh  + (size_t)BS_ * E_;
  _Float16* Wqt = Vt  + (size_t)BS_ * E_;
  _Float16* Wkt = Wqt + (size_t)IN_ * E_;
  _Float16* Wvt = Wkt + (size_t)IN_ * E_;

  prep_w_kernel<<<(3 * IN_ * E_) / 256, 256, 0, stream>>>(Wq, Wk, Wv, Wqt, Wkt, Wvt);
  proj_kernel<<<BS_ / 128, 256, 0, stream>>>(q, Wqt, bq, Qh, 0);
  proj_kernel<<<BS_ / 128, 256, 0, stream>>>(k, Wkt, bk, Kh, 0);
  proj_kernel<<<BS_ / 128, 256, 0, stream>>>(v, Wvt, bv, Vt, 1);

  dim3 grid(S_ / 128, B_);
  flash_attn_kernel<<<grid, 256, 0, stream>>>(Qh, Kh, Vt, out);
}
